// CosineAttention_20392504722137
// MI455X (gfx1250) — matrix operand order repaired and checked by reading, not compiled, not run on hardware
//
#include <hip/hip_runtime.h>

// CosineAttention for MI455X (gfx1250, wave32, WMMA + TDM).
// Pipeline: 3x GEMM (x@Wq/Wk/Wv, bf16 WMMA f32-acc) -> L2-norm+bf16 ->
//           causal flash-attention (bf16 WMMA, TDM K-tile loads) ->
//           GEMM (O@Wo) -> f32 out.

using bf16 = __bf16;
typedef bf16  v16bf __attribute__((ext_vector_type(16)));
typedef bf16  v8bf  __attribute__((ext_vector_type(8)));
typedef float v8f   __attribute__((ext_vector_type(8)));

#ifndef __has_builtin
#define __has_builtin(x) 0
#endif

#if __has_builtin(__builtin_amdgcn_tensor_load_to_lds) && \
    __has_builtin(__builtin_amdgcn_s_wait_tensorcnt)
#define USE_TDM 1
#else
#define USE_TDM 0
#endif

#if defined(__has_include)
#if __has_include(<hip/amd_detail/amd_gfx1250_TDM.h>)
#define TDM_ARITY6 1
#endif
#endif

#if USE_TDM
typedef unsigned int u32x4 __attribute__((ext_vector_type(4)));
typedef int          i32x4 __attribute__((ext_vector_type(4)));
typedef int          i32x8 __attribute__((ext_vector_type(8)));

// Issue a TDM 2D tile load: global (rows x cols bf16, row stride in elems)
// -> LDS at lds_addr, with LDS padding of 4 DWORDs every 32 DWORDs (matches a
// [rows][cols+8] bf16 LDS tile when cols==64). D# packing per CDNA5 ISA ch.8.
__device__ __forceinline__ void tdm_load_tile_bf16(unsigned int lds_addr,
                                                   const void* gaddr,
                                                   unsigned int rows,
                                                   unsigned int cols,
                                                   unsigned int row_stride) {
  const unsigned long long ga = (unsigned long long)(uintptr_t)gaddr;
  u32x4 g0;
  g0[0] = 0x1u;                                   // count=1, user descriptor
  g0[1] = lds_addr;                               // lds_addr [63:32]
  g0[2] = (unsigned int)ga;                       // global_addr [95:64]
  g0[3] = (unsigned int)((ga >> 32) & 0x01FFFFFFu) | (2u << 30);  // type=2
  i32x8 g1;
  // data_size=1 (2B), pad_enable, pad_interval=4 (32 DW), pad_amount=3 (4 DW)
  g1[0] = (int)((1u << 16) | (1u << 20) | (4u << 22) | (3u << 25));
  g1[1] = (int)(cols << 16);                      // tensor_dim0[15:0]
  g1[2] = (int)((cols >> 16) | (rows << 16));     // tdim0[31:16] | tdim1[15:0]
  g1[3] = (int)((rows >> 16) | (cols << 16));     // tdim1[31:16] | tile_dim0
  g1[4] = (int)rows;                              // tile_dim1 (tile_dim2=0)
  g1[5] = (int)row_stride;                        // tensor_dim0_stride[31:0]
  g1[6] = 0;
  g1[7] = 0;                                      // tensor_dim1_stride unused
  i32x4 z4 = {0, 0, 0, 0};                        // groups 2/3: 2D tensor
#if defined(TDM_ARITY6)
  i32x8 z8 = {0, 0, 0, 0, 0, 0, 0, 0};
  __builtin_amdgcn_tensor_load_to_lds(g0, g1, z4, z4, z8, 0);
#else
  __builtin_amdgcn_tensor_load_to_lds(g0, g1, z4, z4, 0);
#endif
}
#endif  // USE_TDM

// ---- fragment helpers (CDNA5 16-bit WMMA VGPR layouts) ----------------------
// A 16x32 (MxK): lane m=lane&15; VGPR0-3 hold K = half*8 + 0..7,
//                VGPR4-7 hold K = 16 + half*8 + 0..7  (half = lane>>4)
// B 32x16 (KxN): lane n=lane&15; same K order as A (half*8 + 0..7, 16 + half*8 + 0..7)
//   -> if B is stored N-major ([n][k]) each lane reads two 8-elem runs.
__device__ __forceinline__ v16bf load_frag16(const bf16* p0, const bf16* p1) {
  v8bf lo = *(const v8bf*)p0;
  v8bf hi = *(const v8bf*)p1;
  v16bf r;
#pragma unroll
  for (int i = 0; i < 8; ++i) { r[i] = lo[i]; r[8 + i] = hi[i]; }
  return r;
}

__device__ __forceinline__ v8f wmma_bf16(v16bf a, v16bf b, v8f c) {
  // (neg_a, A, neg_b, B, c_mod, C, reuse_a, reuse_b)
  return __builtin_amdgcn_wmma_f32_16x16x32_bf16(false, a, false, b, (short)0, c,
                                                 false, false);
}

// Stage 16 elements into 16B-aligned LDS as two b128 stores.
__device__ __forceinline__ void load16_cvt(bf16* dst, const float* src) {
  v8bf a, b;
#pragma unroll
  for (int i = 0; i < 2; ++i) {
    float4 f0 = *(const float4*)(src + 8 * i);
    float4 f1 = *(const float4*)(src + 8 * i + 4);
    v8bf t;
    t[0] = (bf16)f0.x; t[1] = (bf16)f0.y; t[2] = (bf16)f0.z; t[3] = (bf16)f0.w;
    t[4] = (bf16)f1.x; t[5] = (bf16)f1.y; t[6] = (bf16)f1.z; t[7] = (bf16)f1.w;
    if (i == 0) a = t; else b = t;
  }
  *(v8bf*)dst = a;
  *(v8bf*)(dst + 8) = b;
}
__device__ __forceinline__ void load16_cvt(bf16* dst, const bf16* src) {
  *(v8bf*)dst = *(const v8bf*)src;
  *(v8bf*)(dst + 8) = *(const v8bf*)(src + 8);
}

// ---- 128x128 tiled GEMM: C(f32) = A(TA) * Bw(f32), K-step 32 ---------------
// grid = (N/128, M/128), block = 256 threads = 8 waves (4 M x 2 N).
template <typename TA>
__global__ __launch_bounds__(256) void gemm128(const TA* __restrict__ A,
                                               const float* __restrict__ Bw,
                                               float* __restrict__ C,
                                               int K, int lda, int ldb, int ldc) {
  __shared__ bf16 As[128][40];  // [m][k], padded to dodge bank conflicts
  __shared__ bf16 Bs[128][40];  // [n][k] (N-major for contiguous B-frag reads)

  const int tid  = threadIdx.x;
  const int lane = tid & 31, wave = tid >> 5;
  const int half = lane >> 4, l16 = lane & 15;
  const int waveM = wave >> 1, waveN = wave & 1;
  const int row0 = blockIdx.y * 128, n0 = blockIdx.x * 128;

  const v8f zf = {0.f, 0.f, 0.f, 0.f, 0.f, 0.f, 0.f, 0.f};
  v8f acc[2][4];
#pragma unroll
  for (int m = 0; m < 2; ++m)
#pragma unroll
    for (int n = 0; n < 4; ++n) acc[m][n] = zf;

  const int ar = tid >> 1, ac = (tid & 1) * 16;   // A stage: 128 rows x 32 k
  const int br = tid >> 3, bc = (tid & 7) * 16;   // B stage: 32 k x 128 n

  for (int k0 = 0; k0 < K; k0 += 32) {
    // stage A tile (convert to bf16); &As[ar][ac] is 16B aligned (80B rows)
    load16_cvt(&As[ar][ac], A + (size_t)(row0 + ar) * lda + k0 + ac);
    // stage B tile transposed into [n][k]
#pragma unroll
    for (int i = 0; i < 4; ++i) {
      float4 f = *(const float4*)(Bw + (size_t)(k0 + br) * ldb + n0 + bc + 4 * i);
      Bs[bc + 4 * i + 0][br] = (bf16)f.x;
      Bs[bc + 4 * i + 1][br] = (bf16)f.y;
      Bs[bc + 4 * i + 2][br] = (bf16)f.z;
      Bs[bc + 4 * i + 3][br] = (bf16)f.w;
    }
    if (k0 + 32 < K) {  // hint next tiles into cache (global_prefetch_b8)
      __builtin_prefetch(A + (size_t)(row0 + ar) * lda + k0 + 32 + ac, 0, 1);
      __builtin_prefetch(Bw + (size_t)(k0 + 32 + br) * ldb + n0 + bc, 0, 1);
    }
    __syncthreads();

    v16bf afr[2];
#pragma unroll
    for (int mt = 0; mt < 2; ++mt) {
      const int arow = waveM * 32 + mt * 16 + l16;
      afr[mt] = load_frag16(&As[arow][half * 8], &As[arow][16 + half * 8]);
    }
#pragma unroll
    for (int nt = 0; nt < 4; ++nt) {
      const int brow = waveN * 64 + nt * 16 + l16;
      v16bf bfr = load_frag16(&Bs[brow][half * 8], &Bs[brow][16 + half * 8]);
#pragma unroll
      for (int mt = 0; mt < 2; ++mt) acc[mt][nt] = wmma_bf16(afr[mt], bfr, acc[mt][nt]);
    }
    __syncthreads();
  }

  // D layout: VGPR r -> M = r + 8*half, N = l16
#pragma unroll
  for (int mt = 0; mt < 2; ++mt)
#pragma unroll
    for (int nt = 0; nt < 4; ++nt)
#pragma unroll
      for (int r = 0; r < 8; ++r) {
        const int row = row0 + waveM * 32 + mt * 16 + r + 8 * half;
        const int col = n0 + waveN * 64 + nt * 16 + l16;
        C[(size_t)row * ldc + col] = acc[mt][nt][r];
      }
}

// ---- per-head L2 normalize (GROUPS=1) + f32 -> bf16 ------------------------
__global__ __launch_bounds__(256) void norm_convert(const float* __restrict__ src,
                                                    bf16* __restrict__ dst,
                                                    int do_norm) {
  const int gid = blockIdx.x * blockDim.x + threadIdx.x;  // (row, head) pair
  const int row = gid >> 4, h = gid & 15;
  const float* p = src + (size_t)row * 1024 + h * 64;
  float4 v[16];
  float ss = 0.f;
#pragma unroll
  for (int i = 0; i < 16; ++i) {
    v[i] = *(const float4*)(p + 4 * i);
    ss += v[i].x * v[i].x + v[i].y * v[i].y + v[i].z * v[i].z + v[i].w * v[i].w;
  }
  const float sc = do_norm ? rsqrtf(ss + 1e-12f) : 1.0f;
  bf16* q = dst + (size_t)row * 1024 + h * 64;
#pragma unroll
  for (int i = 0; i < 8; ++i) {
    v8bf t;
    t[0] = (bf16)(v[2 * i].x * sc);     t[1] = (bf16)(v[2 * i].y * sc);
    t[2] = (bf16)(v[2 * i].z * sc);     t[3] = (bf16)(v[2 * i].w * sc);
    t[4] = (bf16)(v[2 * i + 1].x * sc); t[5] = (bf16)(v[2 * i + 1].y * sc);
    t[6] = (bf16)(v[2 * i + 1].z * sc); t[7] = (bf16)(v[2 * i + 1].w * sc);
    *(v8bf*)(q + 8 * i) = t;
  }
}

// ---- causal flash attention, cosine-sim (SCALE=8) --------------------------
// grid = (N/64 row-blocks, B*H), block = 128 threads (4 waves x 16 rows).
__global__ __launch_bounds__(128) void flash_attn(const bf16* __restrict__ Q,
                                                  const bf16* __restrict__ Kg,
                                                  const bf16* __restrict__ Vg,
                                                  bf16* __restrict__ O) {
  constexpr int NSEQ = 2048, INNER = 1024;
  __shared__ bf16 Ks[32][72];      // [key][d]   (row-major, padded; TDM pad fits)
  __shared__ bf16 VTs[64][40];     // [d][key]   (transposed, padded)
  __shared__ bf16 Ps[4][16][40];   // per-wave P tile [m][key]

  const int tid  = threadIdx.x;
  const int lane = tid & 31, wave = tid >> 5;
  const int half = lane >> 4, l16 = lane & 15;
  const int bh = blockIdx.y, b = bh >> 4, h = bh & 15;
  const int qrow0 = blockIdx.x * 64;
  const int rbase = qrow0 + wave * 16;

  const bf16* qb = Q + ((size_t)b * NSEQ) * INNER + h * 64;
  const bf16* kb = Kg + ((size_t)b * NSEQ) * INNER + h * 64;
  const bf16* vb = Vg + ((size_t)b * NSEQ) * INNER + h * 64;

  // Q fragments for this wave's 16 rows (hoisted; reused every key block)
  const bf16* qp = qb + (size_t)(rbase + l16) * INNER;
  const v16bf aq0 = load_frag16(qp + half * 8, qp + 16 + half * 8);       // d 0..31
  const v16bf aq1 = load_frag16(qp + 32 + half * 8, qp + 48 + half * 8);  // d 32..63

  const v8f zf = {0.f, 0.f, 0.f, 0.f, 0.f, 0.f, 0.f, 0.f};
  v8f acc[4] = {zf, zf, zf, zf};
  float m_i[8], l_i[8];
#pragma unroll
  for (int r = 0; r < 8; ++r) { m_i[r] = -3.0e38f; l_i[r] = 0.f; }

  const int lr = tid >> 2, lc = (tid & 3) * 16;  // stage: 32 keys x 64 d
  const int nkb = (qrow0 + 64) / 32;

  for (int kbk = 0; kbk < nkb; ++kbk) {
    __syncthreads();
#if USE_TDM
    // K tile via Tensor Data Mover: 32x64 bf16, row stride 1024 elems,
    // LDS-padded to the Ks[32][72] layout. Issued once (wave 0).
    if (wave == 0) {
      tdm_load_tile_bf16((unsigned int)(uintptr_t)&Ks[0][0],
                         kb + (size_t)(kbk * 32) * INNER, 32, 64, INNER);
    }
#else
    {
      const bf16* kp = kb + (size_t)(kbk * 32 + lr) * INNER + lc;
      *(v8bf*)&Ks[lr][lc]     = *(const v8bf*)kp;       // 16B-aligned (144B rows)
      *(v8bf*)&Ks[lr][lc + 8] = *(const v8bf*)(kp + 8);
    }
#endif
    {
      const bf16* vp = vb + (size_t)(kbk * 32 + lr) * INNER + lc;
      v8bf v0v = *(const v8bf*)vp;
      v8bf v1v = *(const v8bf*)(vp + 8);
#pragma unroll
      for (int j = 0; j < 8; ++j) { VTs[lc + j][lr] = v0v[j]; VTs[lc + 8 + j][lr] = v1v[j]; }
    }
#if USE_TDM
    if (wave == 0) __builtin_amdgcn_s_wait_tensorcnt(0);
#endif
    __syncthreads();

    // S = Q * K^T  (two 16x16 n-subtiles, K-dim 64 = 2 WMMA chunks)
    v8f S[2];
#pragma unroll
    for (int nt = 0; nt < 2; ++nt) {
      const int kn = nt * 16 + l16;
      v16bf b0 = load_frag16(&Ks[kn][half * 8], &Ks[kn][16 + half * 8]);
      v16bf b1 = load_frag16(&Ks[kn][32 + half * 8], &Ks[kn][48 + half * 8]);
      v8f s = zf;
      s = wmma_bf16(aq0, b0, s);
      s = wmma_bf16(aq1, b1, s);
      S[nt] = s;
    }

    const bool needmask = (kbk * 32 + 31) > rbase;  // wave-uniform
#pragma unroll
    for (int nt = 0; nt < 2; ++nt)
#pragma unroll
      for (int r = 0; r < 8; ++r) {
        float s = S[nt][r] * 8.0f;  // SCALE
        if (needmask) {
          const int col = kbk * 32 + nt * 16 + l16;
          const int rg  = rbase + r + 8 * half;
          if (col > rg) s = -1e10f;  // MASK_VAL
        }
        S[nt][r] = s;
      }

    // online softmax (row stats reduced across the 16 lanes of each half)
#pragma unroll
    for (int r = 0; r < 8; ++r) {
      float mx = fmaxf(S[0][r], S[1][r]);
#pragma unroll
      for (int o = 1; o < 16; o <<= 1) mx = fmaxf(mx, __shfl_xor(mx, o, 16));
      const float mnew = fmaxf(m_i[r], mx);
      const float sf = __expf(m_i[r] - mnew);
      m_i[r] = mnew;
      l_i[r] *= sf;
#pragma unroll
      for (int t = 0; t < 4; ++t) acc[t][r] *= sf;
      const float p0 = __expf(S[0][r] - mnew);
      const float p1 = __expf(S[1][r] - mnew);
      float rs = p0 + p1;
#pragma unroll
      for (int o = 1; o < 16; o <<= 1) rs += __shfl_xor(rs, o, 16);
      l_i[r] += rs;
      // D-layout -> LDS (wave-private; DS ops are in-order within a wave)
      Ps[wave][r + 8 * half][l16]      = (bf16)p0;
      Ps[wave][r + 8 * half][16 + l16] = (bf16)p1;
    }

    // O += P * V   (A = P 16x32 from LDS in A-layout, B = V^T N-major)
    const v16bf pa = load_frag16(&Ps[wave][l16][half * 8], &Ps[wave][l16][16 + half * 8]);
#pragma unroll
    for (int t = 0; t < 4; ++t) {
      const int vd = t * 16 + l16;
      v16bf bv = load_frag16(&VTs[vd][half * 8], &VTs[vd][16 + half * 8]);
      acc[t] = wmma_bf16(pa, bv, acc[t]);
    }
  }

  // finalize and store O as bf16 at [b, row, h*64 + d]
#pragma unroll
  for (int r = 0; r < 8; ++r) {
    const float inv = 1.0f / l_i[r];
#pragma unroll
    for (int t = 0; t < 4; ++t) acc[t][r] *= inv;
  }
  bf16* op = O + ((size_t)(b * NSEQ + rbase)) * INNER + h * 64;
#pragma unroll
  for (int t = 0; t < 4; ++t)
#pragma unroll
    for (int r = 0; r < 8; ++r)
      op[(size_t)(r + 8 * half) * INNER + t * 16 + l16] = (bf16)acc[t][r];
}

// ---- launcher ---------------------------------------------------------------
extern "C" void kernel_launch(void* const* d_in, const int* in_sizes, int n_in,
                              void* d_out, int out_size, void* d_ws, size_t ws_size,
                              hipStream_t stream) {
  (void)in_sizes; (void)n_in; (void)out_size; (void)ws_size;
  const float* x  = (const float*)d_in[0];
  const float* Wq = (const float*)d_in[1];
  const float* Wk = (const float*)d_in[2];
  const float* Wv = (const float*)d_in[3];
  const float* Wo = (const float*)d_in[4];
  float* out = (float*)d_out;

  const size_t SZ = (size_t)4096 * 1024;  // B*N x INNER
  float* Qf = (float*)d_ws;
  float* Kf = Qf + SZ;
  float* Vf = Kf + SZ;
  bf16* Qn = (bf16*)(Vf + SZ);
  bf16* Kn = Qn + SZ;
  bf16* Vn = Kn + SZ;
  bf16* On = Vn + SZ;

  dim3 gg(8, 32);  // N/128, M/128
  gemm128<float><<<gg, 256, 0, stream>>>(x, Wq, Qf, 1024, 1024, 1024, 1024);
  gemm128<float><<<gg, 256, 0, stream>>>(x, Wk, Kf, 1024, 1024, 1024, 1024);
  gemm128<float><<<gg, 256, 0, stream>>>(x, Wv, Vf, 1024, 1024, 1024, 1024);

  norm_convert<<<256, 256, 0, stream>>>(Qf, Qn, 1);
  norm_convert<<<256, 256, 0, stream>>>(Kf, Kn, 1);
  norm_convert<<<256, 256, 0, stream>>>(Vf, Vn, 0);

  flash_attn<<<dim3(32, 32), 128, 0, stream>>>(Qn, Kn, Vn, On);

  gemm128<bf16><<<gg, 256, 0, stream>>>(On, Wo, out, 1024, 1024, 1024, 1024);
}
